// GlobalPointer_6614249636505
// MI455X (gfx1250) — compile-verified, matrix operand order checked by reading, not run on hardware
//
#include <hip/hip_runtime.h>
#include <math.h>

#define B_ 16
#define L_ 512
#define D_ 768
#define T_ 12
#define A_ 64
#define N_ (T_ * A_ * 2)   // 1536 = 12 heads x 128
#define M_ (B_ * L_)       // 8192

typedef __attribute__((ext_vector_type(16))) __bf16 v16bf;
typedef __attribute__((ext_vector_type(8)))  float  v8f;

union FragU { v16bf v; unsigned u[8]; };

__device__ __forceinline__ unsigned short f32_to_bf16(float f) {
  unsigned x = __float_as_uint(f);
  x += 0x7FFFu + ((x >> 16) & 1u);   // round-to-nearest-even
  return (unsigned short)(x >> 16);
}

__device__ __forceinline__ float bf16_to_f32(unsigned short h) {
  return __uint_as_float(((unsigned)h) << 16);
}

// Load one 16x32 bf16 operand fragment from an LDS tile laid out row-major
// [row][k] with element stride `ld`. ISA 7.12.2 16-bit A layout:
// lanes 0-15: K = 0..7 (V0-3), 16..23 (V4-7); lanes 16-31: K = 8..15, 24..31.
__device__ __forceinline__ v16bf lds_frag(const unsigned short* tile, int row0,
                                          int ld, int k0) {
  const int lane = threadIdx.x & 31;
  const int m  = row0 + (lane & 15);
  const int hi = (lane >> 4) << 3;           // 0 or 8
  const unsigned* p = (const unsigned*)tile;
  const int b0 = m * ld + k0 + hi;
  FragU f;
#pragma unroll
  for (int v = 0; v < 4; ++v) {
    f.u[v]     = p[(b0 + 2 * v) >> 1];
    f.u[v + 4] = p[(b0 + 16 + 2 * v) >> 1];
  }
  return f.v;
}

// ---------------- prep: f32 -> bf16 copy of the activation matrix -----------
__global__ __launch_bounds__(256) void cvt_bf16_kernel(
    const float* __restrict__ src, unsigned short* __restrict__ dst, int n) {
  int i = blockIdx.x * 256 + threadIdx.x;
  if (i < n) dst[i] = f32_to_bf16(src[i]);
}

// ---------------- prep: W [D,N] f32 -> Wt [N,D] bf16 ------------------------
__global__ __launch_bounds__(256) void wt_kernel(
    const float* __restrict__ W, unsigned short* __restrict__ Wt) {
  int i = blockIdx.x * 256 + threadIdx.x;
  if (i >= N_ * D_) return;
  int kd = i % D_;
  int n  = i / D_;
  Wt[n * D_ + kd] = f32_to_bf16(W[kd * N_ + n]);
}

// ------- GEMM1 fused with bias + RoPE: one head (128 N-cols) per block ------
// Block tile 64(M) x 128(N); 8 wave32s in 2(M) x 4(N); each wave 32x32.
// K-slab of 64 per LDS fill; next slab preloaded into registers while the
// current slab's 8 WMMAs run (register double-buffering). Epilogue: accum ->
// bf16 in LDS, then block-local RoPE writes qb/kb in [b*T+t][l][a] layout.
__global__ __launch_bounds__(256) void gemm1_rope_kernel(
    const unsigned short* __restrict__ Xb, const unsigned short* __restrict__ Wt,
    const float* __restrict__ bias,
    unsigned short* __restrict__ qb, unsigned short* __restrict__ kb) {
  __shared__ unsigned short smem[(64 + 128) * 72];  // 27.6 KB; epilogue reuses
  unsigned short* sA = smem;                        // 64 x 72
  unsigned short* sB = smem + 64 * 72;              // 128 x 72

  const int t   = blockIdx.x;         // head index, N range [t*128, t*128+128)
  const int n0  = t * 128;
  const int m0  = blockIdx.y * 64;
  const int tid = threadIdx.x;
  const int rA  = tid >> 2;           // 0..63
  const int cA  = (tid & 3) * 16;     // 2 x uint4
  const int rB  = tid >> 1;           // 0..127
  const int cB  = (tid & 1) * 32;     // 4 x uint4
  const int wid = tid >> 5;
  const int wm  = wid & 1;            // 2 M-blocks of 32
  const int wn  = wid >> 1;           // 4 N-blocks of 32
  const int lane = tid & 31;

  const unsigned short* gA = Xb + (m0 + rA) * D_ + cA;
  const unsigned short* gB = Wt + (n0 + rB) * D_ + cB;
  unsigned short* lA = &sA[rA * 72 + cA];
  unsigned short* lB = &sB[rB * 72 + cB];

  uint4 ra0, ra1, rb0, rb1, rb2, rb3;          // register stage for next slab
  auto preload = [&](int k0) {
    ra0 = *(const uint4*)(gA + k0);
    ra1 = *(const uint4*)(gA + k0 + 8);
    rb0 = *(const uint4*)(gB + k0);
    rb1 = *(const uint4*)(gB + k0 + 8);
    rb2 = *(const uint4*)(gB + k0 + 16);
    rb3 = *(const uint4*)(gB + k0 + 24);
  };
  preload(0);

  v8f c00 = {}, c01 = {}, c10 = {}, c11 = {};
  for (int k0 = 0; k0 < D_; k0 += 64) {
    __syncthreads();                  // prior slab's LDS reads complete
    *(uint4*)lA       = ra0;
    *(uint4*)(lA + 8) = ra1;
    *(uint4*)lB        = rb0;
    *(uint4*)(lB + 8)  = rb1;
    *(uint4*)(lB + 16) = rb2;
    *(uint4*)(lB + 24) = rb3;
    if (k0 + 64 < D_) {
      preload(k0 + 64);               // overlaps the WMMA phase below
      if (k0 + 128 < D_) {            // warm GL2 two slabs ahead
        __builtin_prefetch((const void*)(gA + k0 + 128), 0, 3);
        __builtin_prefetch((const void*)(gB + k0 + 128), 0, 3);
      }
    }
    __syncthreads();
#pragma unroll
    for (int kk = 0; kk < 64; kk += 32) {
      v16bf a0 = lds_frag(sA, wm * 32, 72, kk);
      v16bf a1 = lds_frag(sA, wm * 32 + 16, 72, kk);
      v16bf b0 = lds_frag(sB, wn * 32, 72, kk);
      v16bf b1 = lds_frag(sB, wn * 32 + 16, 72, kk);
      c00 = __builtin_amdgcn_wmma_f32_16x16x32_bf16(false, a0, false, b0, (short)0, c00, false, false);
      c01 = __builtin_amdgcn_wmma_f32_16x16x32_bf16(false, a0, false, b1, (short)0, c01, false, false);
      c10 = __builtin_amdgcn_wmma_f32_16x16x32_bf16(false, a1, false, b0, (short)0, c10, false, false);
      c11 = __builtin_amdgcn_wmma_f32_16x16x32_bf16(false, a1, false, b1, (short)0, c11, false, false);
    }
  }
  __syncthreads();                    // last slab's LDS reads complete

  // ---- epilogue 1: accumulators (+bias) -> bf16 tile xt[64][136] in LDS ----
  unsigned short* xt = smem;   // safe: barrier above ended all tile reads
  const int hi8 = (lane >> 4) << 3;
  const int cN  = lane & 15;
  {
    const int col0 = wn * 32 + cN;         // local N col for sn=0
    const int col1 = col0 + 16;            // sn=1
    const float bias0 = bias[n0 + col0];
    const float bias1 = bias[n0 + col1];
#pragma unroll
    for (int rr = 0; rr < 8; ++rr) {
      int r0 = wm * 32 + hi8 + rr;         // sm=0
      int r1 = r0 + 16;                    // sm=1
      xt[r0 * 136 + col0] = f32_to_bf16(c00[rr] + bias0);
      xt[r0 * 136 + col1] = f32_to_bf16(c01[rr] + bias1);
      xt[r1 * 136 + col0] = f32_to_bf16(c10[rr] + bias0);
      xt[r1 * 136 + col1] = f32_to_bf16(c11[rr] + bias1);
    }
  }
  __syncthreads();

  // ---- epilogue 2: RoPE within the head, write q/k bf16 [bt][l][a] ---------
  // a = (tid + 256*s) & 63 == tid & 63 : all a-dependent math hoisted.
  {
    const int a    = tid & 63;
    const int rowb = tid >> 6;             // 0..3; row = rowb + 4*s
    int ap; float sgn;
    if (a < 32) { ap = 2 * a + 1;     sgn = -1.f; }
    else        { ap = 2 * (a - 32);  sgn =  1.f; }
    // theta = 10000^(-(a>>1)/32) = exp2(-(a>>1) * log2(10000)/32)
    const float theta = exp2f(-(float)(a >> 1) * 0.41524101186092029f);
#pragma unroll
    for (int s = 0; s < 16; ++s) {
      int row = rowb + 4 * s;              // 0..63
      int m   = m0 + row;
      int b   = m >> 9;                    // m / 512
      int l   = m & 511;
      unsigned qk  = *(const unsigned*)&xt[row * 136 + 2 * a];   // lo=q, hi=k
      unsigned qkp = *(const unsigned*)&xt[row * 136 + 2 * ap];
      float q  = bf16_to_f32((unsigned short)(qk & 0xFFFFu));
      float k  = bf16_to_f32((unsigned short)(qk >> 16));
      float qp = sgn * bf16_to_f32((unsigned short)(qkp & 0xFFFFu));
      float kp = sgn * bf16_to_f32((unsigned short)(qkp >> 16));
      float pe = (float)l * theta;
      float sn = __sinf(pe);
      float cs = __cosf(pe);
      size_t o = ((size_t)(b * T_ + t) * L_ + l) * A_ + a;
      qb[o] = f32_to_bf16(q * cs + qp * sn);
      kb[o] = f32_to_bf16(k * cs + kp * sn);
    }
  }
}

// ---------------- attention scores + fused pad/triu mask --------------------
// Per (b,t): S[512,512] = q[512,64] * k[512,64]^T ; block tile 128(i) x 64(j),
// 8 wave32s in 4(i) x 2(j), each wave 32x32 = 8 WMMAs over K=64.
__global__ __launch_bounds__(256) void attn_kernel(
    const unsigned short* __restrict__ qb, const unsigned short* __restrict__ kb,
    const int* __restrict__ mask, float* __restrict__ out) {
  __shared__ unsigned short sQ[128 * 72];
  __shared__ unsigned short sK[64 * 72];

  const int j0  = blockIdx.x * 64;
  const int i0  = blockIdx.y * 128;
  const int bt  = blockIdx.z;
  const int b   = bt / T_;
  const int tid = threadIdx.x;
  const int rQ  = tid >> 1;          // 0..127
  const int cQ  = (tid & 1) * 32;
  const int rK  = tid >> 2;          // 0..63
  const int cK  = (tid & 3) * 16;
  const int wid = tid >> 5;
  const int wm  = wid >> 1;          // 4 i-blocks of 32
  const int wn  = wid & 1;           // 2 j-blocks of 32
  const int lane = tid & 31;

  const unsigned short* gq = qb + ((size_t)bt * L_ + i0 + rQ) * A_ + cQ;
  const unsigned short* gk = kb + ((size_t)bt * L_ + j0 + rK) * A_ + cK;
#pragma unroll
  for (int u = 0; u < 4; ++u)
    *(uint4*)&sQ[rQ * 72 + cQ + 8 * u] = *(const uint4*)(gq + 8 * u);
  *(uint4*)&sK[rK * 72 + cK]     = *(const uint4*)(gk);
  *(uint4*)&sK[rK * 72 + cK + 8] = *(const uint4*)(gk + 8);
  __syncthreads();

  v8f c00 = {}, c01 = {}, c10 = {}, c11 = {};
#pragma unroll
  for (int k0 = 0; k0 < A_; k0 += 32) {
    v16bf a0 = lds_frag(sQ, wm * 32, 72, k0);
    v16bf a1 = lds_frag(sQ, wm * 32 + 16, 72, k0);
    v16bf b0 = lds_frag(sK, wn * 32, 72, k0);
    v16bf b1 = lds_frag(sK, wn * 32 + 16, 72, k0);
    c00 = __builtin_amdgcn_wmma_f32_16x16x32_bf16(false, a0, false, b0, (short)0, c00, false, false);
    c01 = __builtin_amdgcn_wmma_f32_16x16x32_bf16(false, a0, false, b1, (short)0, c01, false, false);
    c10 = __builtin_amdgcn_wmma_f32_16x16x32_bf16(false, a1, false, b0, (short)0, c10, false, false);
    c11 = __builtin_amdgcn_wmma_f32_16x16x32_bf16(false, a1, false, b1, (short)0, c11, false, false);
  }

  const int hi8 = (lane >> 4) << 3;
  const int cN  = lane & 15;
  const int j_0 = j0 + wn * 32 + cN;
  const int j_1 = j_0 + 16;
  const bool mj0 = mask[b * L_ + j_0] > 0;
  const bool mj1 = mask[b * L_ + j_1] > 0;
  float* obase = out + (size_t)bt * L_ * L_;
#pragma unroll
  for (int rr = 0; rr < 8; ++rr) {
    int i_0 = i0 + wm * 32 + hi8 + rr;       // sm=0
    int i_1 = i_0 + 16;                       // sm=1
    bool mi0 = mask[b * L_ + i_0] > 0;
    bool mi1 = mask[b * L_ + i_1] > 0;
    obase[(size_t)i_0 * L_ + j_0] = (mi0 && mj0 && (j_0 >= i_0)) ? c00[rr] : -1e12f;
    obase[(size_t)i_0 * L_ + j_1] = (mi0 && mj1 && (j_1 >= i_0)) ? c01[rr] : -1e12f;
    obase[(size_t)i_1 * L_ + j_0] = (mi1 && mj0 && (j_0 >= i_1)) ? c10[rr] : -1e12f;
    obase[(size_t)i_1 * L_ + j_1] = (mi1 && mj1 && (j_1 >= i_1)) ? c11[rr] : -1e12f;
  }
}

extern "C" void kernel_launch(void* const* d_in, const int* in_sizes, int n_in,
                              void* d_out, int out_size, void* d_ws, size_t ws_size,
                              hipStream_t stream) {
  const float* x_in = (const float*)d_in[0];   // [B,L,D]
  const int*   mask = (const int*)d_in[1];     // [B,L]
  const float* W    = (const float*)d_in[2];   // [D,N]
  const float* bias = (const float*)d_in[3];   // [N]
  float*       out  = (float*)d_out;           // [B,T,L,L]

  char* ws = (char*)d_ws;
  unsigned short* Xb = (unsigned short*)ws; ws += (size_t)M_ * D_ * 2;            // 12.6 MB
  unsigned short* Wt = (unsigned short*)ws; ws += (size_t)N_ * D_ * 2;            //  2.4 MB
  unsigned short* qb = (unsigned short*)ws; ws += (size_t)B_ * T_ * L_ * A_ * 2;  // 12.6 MB
  unsigned short* kb = (unsigned short*)ws;                                       // 12.6 MB

  {
    int n = M_ * D_;
    cvt_bf16_kernel<<<(n + 255) / 256, 256, 0, stream>>>(x_in, Xb, n);
  }
  {
    int n = N_ * D_;
    wt_kernel<<<(n + 255) / 256, 256, 0, stream>>>(W, Wt);
  }
  gemm1_rope_kernel<<<dim3(T_, M_ / 64), 256, 0, stream>>>(Xb, Wt, bias, qb, kb);
  attn_kernel<<<dim3(L_ / 64, L_ / 128, B_ * T_), 256, 0, stream>>>(qb, kb, mask, out);
}
